// MyNet_51213190037930
// MI455X (gfx1250) — compile-verified, matrix operand order checked by reading, not run on hardware
//
#include <hip/hip_runtime.h>
#include <hip/hip_bf16.h>

typedef __attribute__((ext_vector_type(16))) _Float16 v16h;
typedef __attribute__((ext_vector_type(8)))  _Float16 v8h;
typedef __attribute__((ext_vector_type(8)))  float    v8f;

#define NIMG 4096

// ---------------- utility kernels ----------------

__global__ void zero_kernel(float* p, int n) {
    int i = blockIdx.x * blockDim.x + threadIdx.x;
    if (i < n) p[i] = 0.0f;
}

// Repack conv weight OIHW f32 -> [Cout][KS][KS][Cin] f16  (K-major, channel fastest)
__global__ void repack_conv_w_kernel(const float* __restrict__ src, _Float16* __restrict__ dst,
                                     int Cout, int Cin, int KS) {
    int total = Cout * Cin * KS * KS;
    int i = blockIdx.x * blockDim.x + threadIdx.x;   // dst index
    if (i >= total) return;
    int ci = i % Cin;      int t = i / Cin;
    int kw = t % KS;       t /= KS;
    int kh = t % KS;       int o = t / KS;
    dst[i] = (_Float16)src[((o * Cin + ci) * KS + kh) * KS + kw];
}

// lw1 (128,1024) with k = c*16 + h*4 + w  ->  f16 rows with k' = (h*4+w)*64 + c (NHWC flatten)
__global__ void repack_lw1_kernel(const float* __restrict__ src, _Float16* __restrict__ dst) {
    int i = blockIdx.x * blockDim.x + threadIdx.x;   // dst index, 128*1024
    if (i >= 128 * 1024) return;
    int o  = i >> 10;
    int r  = i & 1023;
    int c  = r & 63;
    int hw = r >> 6;
    dst[i] = (_Float16)src[o * 1024 + c * 16 + hw];
}

// conv0: 1x1 conv 1->32, NHWC output y[(n*784+p)*32 + c]
__global__ void conv0_kernel(const float* __restrict__ x, const float* __restrict__ w,
                             float* __restrict__ y) {
    const long total = (long)NIMG * 784 * 32;
    for (long i = blockIdx.x * (long)blockDim.x + threadIdx.x; i < total;
         i += (long)gridDim.x * blockDim.x) {
        int  c = (int)(i & 31);
        long p = i >> 5;                 // n*784 + pixel
        y[i] = x[p] * w[c];
    }
}

// ---------------- implicit-GEMM conv via WMMA (NHWC, K-major weights) ----------------
// act: [N][IH][IW][Cin] f16, wgt: [Cout][KS*KS*Cin] f16, y: [N][OH][OW][Cout] f32.
// One wave: 32(M) x 16(Cout) tile = two 16x16 accumulators sharing B.
// All fragment halves are contiguous 16B chunks -> global_load_b128.
template<int Cin, int KS, int STRIDE, int IH, int IW, int OH, int OW, int Cout>
__global__ __launch_bounds__(256)
void conv_wmma_kernel(const _Float16* __restrict__ act, const _Float16* __restrict__ wgt,
                      float* __restrict__ y) {
    constexpr int K      = Cin * KS * KS;
    constexpr int Mtot   = NIMG * OH * OW;          // always a multiple of 32
    constexpr int tilesC = Cout / 16;
    constexpr int tilesM = Mtot / 32;

    const int lane = threadIdx.x & 31;
    const int wave = threadIdx.x >> 5;
    int tile = blockIdx.x * 8 + wave;               // wave-uniform
    if (tile >= tilesM * tilesC) return;            // EXEC stays all-1s per wave
    const int tile_c = tile % tilesC;
    const int tile_m = tile / tilesC;
    const int half = lane >> 4;
    const int l16  = lane & 15;

    // two A rows per lane: m0 and m0+16
    int ih0[2], iw0[2];
    const _Float16* abase[2];
    {
        int m = tile_m * 32 + l16;
#pragma unroll
        for (int s = 0; s < 2; ++s, m += 16) {
            int ow = m % OW; int t = m / OW;
            int oh = t % OH; int n = t / OH;
            ih0[s] = oh * STRIDE; iw0[s] = ow * STRIDE;
            abase[s] = act + (size_t)n * IH * IW * Cin;
        }
    }
    const _Float16* brow = wgt + (size_t)(tile_c * 16 + l16) * K;

    auto achunk = [&](int s, int k0) -> v8h {       // k0 is a multiple of 8, chunk within one (kh,kw) run
        int rc = k0 / Cin;                          // kh*KS + kw   (Cin is a power of two)
        int ci = k0 - rc * Cin;
        int kh = rc / KS;
        int kw = rc - kh * KS;
        return *(const v8h*)(abase[s] + ((size_t)(ih0[s] + kh) * IW + (iw0[s] + kw)) * Cin + ci);
    };

    v8f c0 = {}, c1 = {};
#pragma unroll 4
    for (int kt = 0; kt < K; kt += 32) {
        const int kA = kt + half * 8;
        v8h a0l = achunk(0, kA), a0h = achunk(0, kA + 16);
        v8h a1l = achunk(1, kA), a1h = achunk(1, kA + 16);
        v8h bl  = *(const v8h*)(brow + kA);
        v8h bh  = *(const v8h*)(brow + kA + 16);
        v16h A0 = __builtin_shufflevector(a0l, a0h, 0,1,2,3,4,5,6,7,8,9,10,11,12,13,14,15);
        v16h A1 = __builtin_shufflevector(a1l, a1h, 0,1,2,3,4,5,6,7,8,9,10,11,12,13,14,15);
        v16h B  = __builtin_shufflevector(bl,  bh,  0,1,2,3,4,5,6,7,8,9,10,11,12,13,14,15);
        c0 = __builtin_amdgcn_wmma_f32_16x16x32_f16(false, A0, false, B, (short)0, c0, false, false);
        c1 = __builtin_amdgcn_wmma_f32_16x16x32_f16(false, A1, false, B, (short)0, c1, false, false);
    }

    // store NHWC: column (= cout) contiguous across lanes -> coalesced b32 stores
    const int coutD = tile_c * 16 + l16;
#pragma unroll
    for (int r = 0; r < 8; ++r) {
        int md = tile_m * 32 + r + half * 8;
        y[(size_t)md * Cout + coutD]        = c0[r];
        y[(size_t)(md + 16) * Cout + coutD] = c1[r];
    }
}

// ---------------- BatchNorm (batch stats) + PReLU, NHWC ----------------
// y[row*C + c], rows = N*HW. blockDim=256, C divides 256.
__global__ void bn_stats_kernel(const float* __restrict__ y, float* __restrict__ sum,
                                float* __restrict__ sumsq, int C, long rows) {
    __shared__ float ss[256], sq[256];
    const int c = threadIdx.x % C;
    const int g = threadIdx.x / C;
    const int G = blockDim.x / C;
    float s = 0.0f, q = 0.0f;
    for (long r = (long)blockIdx.x * G + g; r < rows; r += (long)gridDim.x * G) {
        float v = y[r * C + c];
        s += v; q += v * v;
    }
    ss[threadIdx.x] = s; sq[threadIdx.x] = q;
    __syncthreads();
    if (g == 0) {
        for (int gg = 1; gg < G; ++gg) { s += ss[gg * C + c]; q += sq[gg * C + c]; }
        atomicAdd(&sum[c], s); atomicAdd(&sumsq[c], q);
    }
}

__global__ void bn_finalize_kernel(const float* sum, const float* sumsq,
                                   const float* g, const float* b,
                                   float* scale, float* shift, int C, float cnt) {
    int c = threadIdx.x;
    if (c >= C) return;
    float mean = sum[c] / cnt;
    float var  = sumsq[c] / cnt - mean * mean;
    float sc   = g[c] * rsqrtf(var + 1e-5f);
    scale[c] = sc;
    shift[c] = b[c] - mean * sc;
}

__global__ void bn_apply_prelu_kernel(const float* __restrict__ y,
                                      const float* __restrict__ scale,
                                      const float* __restrict__ shift,
                                      const float* __restrict__ alpha_p,
                                      _Float16* __restrict__ out,
                                      long total, int Cmask) {     // C power of two -> mask
    const float alpha = *alpha_p;
    for (long i = blockIdx.x * (long)blockDim.x + threadIdx.x; i < total;
         i += (long)gridDim.x * blockDim.x) {
        int c = (int)(i & Cmask);
        float v = y[i] * scale[c] + shift[c];
        out[i] = (_Float16)(v > 0.0f ? v : alpha * v);
    }
}

// ---------------- tail: FC2 + ArcLoss ----------------

__global__ void fc2_kernel(const _Float16* __restrict__ z, const float* __restrict__ w2,
                           const float* __restrict__ b2, float* __restrict__ feat) {
    int i = blockIdx.x * blockDim.x + threadIdx.x;
    if (i >= NIMG * 2) return;
    int o = i & 1, n = i >> 1;
    float acc = b2[o];
    for (int k = 0; k < 128; ++k) acc += (float)z[n * 128 + k] * w2[o * 128 + k];
    feat[n * 2 + o] = acc;
}

__global__ void arc_kernel(const float* __restrict__ feat, const int* __restrict__ labels,
                           const float* __restrict__ aw, float* __restrict__ wf_out,
                           float* __restrict__ loss_acc) {
    int n = blockIdx.x * blockDim.x + threadIdx.x;
    if (n >= NIMG) return;
    const float S = 64.0f, Mm = 0.5f, EPS = 1e-7f;
    float f0 = feat[2 * n], f1 = feat[2 * n + 1];
    float inv = rsqrtf(f0 * f0 + f1 * f1);
    float x0 = f0 * inv, x1 = f1 * inv;
    float wf[10];
    float sumexp = 0.0f;
#pragma unroll
    for (int j = 0; j < 10; ++j) {
        wf[j] = x0 * aw[2 * j] + x1 * aw[2 * j + 1];
        wf_out[n * 10 + j] = wf[j];
        sumexp += __expf(S * wf[j]);
    }
    int lab = labels[n];
    float tgt = wf[lab];
    float cc  = fminf(fmaxf(tgt, -1.0f + EPS), 1.0f - EPS);
    float num = S * __cosf(acosf(cc) + Mm);
    float den = __expf(num) + sumexp - __expf(S * tgt);
    atomicAdd(loss_acc, num - __logf(den));
}

__global__ void loss_final_kernel(const float* acc, float* out) {
    out[0] = -(acc[0] / (float)NIMG);
}

// ---------------- host orchestration ----------------

extern "C" void kernel_launch(void* const* d_in, const int* in_sizes, int n_in,
                              void* d_out, int out_size, void* d_ws, size_t ws_size,
                              hipStream_t stream) {
    const float* x      = (const float*)d_in[0];
    const int*   labels = (const int*)  d_in[1];
    const float* cw[6]; const float* bg[6]; const float* bb[6]; const float* pa[6];
    for (int i = 0; i < 6; ++i) {
        cw[i] = (const float*)d_in[2 + 4 * i];
        bg[i] = (const float*)d_in[3 + 4 * i];
        bb[i] = (const float*)d_in[4 + 4 * i];
        pa[i] = (const float*)d_in[5 + 4 * i];
    }
    const float* lw1 = (const float*)d_in[26];
    const float* lg1 = (const float*)d_in[27];
    const float* lb1 = (const float*)d_in[28];
    const float* pl1 = (const float*)d_in[29];
    const float* lw2 = (const float*)d_in[30];
    const float* lb2 = (const float*)d_in[31];
    const float* aw  = (const float*)d_in[32];

    // workspace layout
    char* ws = (char*)d_ws;
    _Float16* w16   = (_Float16*)ws;                         // repacked f16 weights
    float*    stats = (float*)(ws + 524288);
    float*    sum = stats, *sumsq = stats + 128, *scale = stats + 256, *shift = stats + 384;
    float*    lossacc = stats + 512;
    float*    ybuf = (float*)(ws + 524288 + 4096);           // raw conv out (NHWC), cap 4096*784*32 f32
    _Float16* actA = (_Float16*)(ws + 528384 + 411041792ll);
    _Float16* actB = (_Float16*)(ws + 528384 + 411041792ll + 205520896ll);

    float* feat = (float*)d_out;
    float* wf   = feat + 2 * NIMG;
    float* loss = feat + 12 * NIMG;

    // f16 weight offsets (elements)
    const int W1 = 0, W2 = 18432, W3 = 20480, W4 = 38912, W5 = 40960, LW1 = 59392;
    repack_conv_w_kernel<<<(18432 + 255) / 256, 256, 0, stream>>>(cw[1], w16 + W1, 64, 32, 3);
    repack_conv_w_kernel<<<(2048  + 255) / 256, 256, 0, stream>>>(cw[2], w16 + W2, 32, 64, 1);
    repack_conv_w_kernel<<<(18432 + 255) / 256, 256, 0, stream>>>(cw[3], w16 + W3, 64, 32, 3);
    repack_conv_w_kernel<<<(2048  + 255) / 256, 256, 0, stream>>>(cw[4], w16 + W4, 32, 64, 1);
    repack_conv_w_kernel<<<(18432 + 255) / 256, 256, 0, stream>>>(cw[5], w16 + W5, 64, 32, 3);
    repack_lw1_kernel<<<(131072 + 255) / 256, 256, 0, stream>>>(lw1, w16 + LW1);

    auto bn = [&](const float* g, const float* b, const float* alpha,
                  int C, int HW, _Float16* outAct) {
        long rows = (long)NIMG * HW;
        zero_kernel<<<1, 256, 0, stream>>>(stats, 256);
        bn_stats_kernel<<<256, 256, 0, stream>>>(ybuf, sum, sumsq, C, rows);
        bn_finalize_kernel<<<1, 128, 0, stream>>>(sum, sumsq, g, b, scale, shift, C, (float)rows);
        bn_apply_prelu_kernel<<<2048, 256, 0, stream>>>(ybuf, scale, shift, alpha,
                                                        outAct, rows * C, C - 1);
    };

#define LAUNCH_CONV(CIN, KS, ST, IH, IW, OH, OW, COUT, IN, WOFF)                          \
    do {                                                                                  \
        constexpr int tiles_ = (NIMG * (OH) * (OW) / 32) * ((COUT) / 16);                 \
        conv_wmma_kernel<CIN, KS, ST, IH, IW, OH, OW, COUT>                               \
            <<<(tiles_ + 7) / 8, 256, 0, stream>>>(IN, w16 + (WOFF), ybuf);               \
    } while (0)

    // L0: 1x1 conv 1->32 (scalar) -> NHWC [N,28,28,32]
    conv0_kernel<<<4096, 256, 0, stream>>>(x, cw[0], ybuf);
    bn(bg[0], bb[0], pa[0], 32, 784, actA);
    // L1: 3x3 s2 32->64 -> [N,13,13,64]
    LAUNCH_CONV(32, 3, 2, 28, 28, 13, 13, 64, actA, W1);   bn(bg[1], bb[1], pa[1], 64, 169, actB);
    // L2: 1x1 64->32
    LAUNCH_CONV(64, 1, 1, 13, 13, 13, 13, 32, actB, W2);   bn(bg[2], bb[2], pa[2], 32, 169, actA);
    // L3: 3x3 s2 32->64 -> [N,6,6,64]
    LAUNCH_CONV(32, 3, 2, 13, 13, 6, 6, 64, actA, W3);     bn(bg[3], bb[3], pa[3], 64, 36, actB);
    // L4: 1x1 64->32
    LAUNCH_CONV(64, 1, 1, 6, 6, 6, 6, 32, actB, W4);       bn(bg[4], bb[4], pa[4], 32, 36, actA);
    // L5: 3x3 s1 32->64 -> [N,4,4,64]
    LAUNCH_CONV(32, 3, 1, 6, 6, 4, 4, 64, actA, W5);       bn(bg[5], bb[5], pa[5], 64, 16, actB);
    // FC1 as 1x1 conv over [N,1,1,1024] (lw1 repacked to NHWC flatten order)
    LAUNCH_CONV(1024, 1, 1, 1, 1, 1, 1, 128, actB, LW1);   bn(lg1, lb1, pl1, 128, 1, actA);
    // FC2 -> features
    fc2_kernel<<<(NIMG * 2 + 255) / 256, 256, 0, stream>>>(actA, lw2, lb2, feat);
    // ArcLoss
    zero_kernel<<<1, 32, 0, stream>>>(lossacc, 1);
    arc_kernel<<<(NIMG + 255) / 256, 256, 0, stream>>>(feat, labels, aw, wf, lossacc);
    loss_final_kernel<<<1, 1, 0, stream>>>(lossacc, loss);
#undef LAUNCH_CONV
}
